// MultiHead_Attention_64725157151331
// MI455X (gfx1250) — compile-verified, matrix operand order checked by reading, not run on hardware
//
#include <hip/hip_runtime.h>
#include <hip/hip_bf16.h>

typedef __attribute__((ext_vector_type(16))) __bf16 v16bf;
typedef __attribute__((ext_vector_type(8)))  float  v8f;

union FragAB  { v16bf v; uint4 u[2]; };
union FragAcc { v8f   v; uint4 u[2]; float f[8]; };

#define SCALE 0.125f   // 64^-0.5

static __device__ __forceinline__ unsigned short f32_to_bf16(float f) {
    union { float f; unsigned u; } v; v.f = f;
    unsigned u = v.u;
    u += 0x7FFFu + ((u >> 16) & 1u);   // round-to-nearest-even
    return (unsigned short)(u >> 16);
}

static __device__ __forceinline__ v8f wmma_bf16(const FragAB& a, const FragAB& b, v8f c) {
    return __builtin_amdgcn_wmma_f32_16x16x32_bf16(false, a.v, false, b.v,
                                                   (short)0, c, false, false);
}

// Load one 16x32 A (or 32x16 B) bf16 fragment: lane holds M/N = mr,
// K halves {8g+0..7} and {16+8g+0..7}; row-major source, K contiguous.
static __device__ __forceinline__ void load_frag(FragAB& f,
                                                 const unsigned short* __restrict__ row,
                                                 int off, int g) {
    f.u[0] = *(const uint4*)(row + off +      g * 8);
    f.u[1] = *(const uint4*)(row + off + 16 + g * 8);
}

// ---------------- pass 0: converts ----------------

__global__ void convert_x_kernel(const float* __restrict__ in,
                                 unsigned short* __restrict__ out, int n) {
    int i = blockIdx.x * blockDim.x + threadIdx.x;
    if (i < n) out[i] = f32_to_bf16(in[i]);
}

// in: fp32 [K][N] -> out: bf16 [N][K]
__global__ void transpose_convert_kernel(const float* __restrict__ in,
                                         unsigned short* __restrict__ out,
                                         int K, int N) {
    __shared__ float tile[32][33];
    int kBase = blockIdx.y * 32, nBase = blockIdx.x * 32;
    for (int i = threadIdx.y; i < 32; i += 8) {
        tile[i][threadIdx.x] = in[(kBase + i) * N + (nBase + threadIdx.x)];
    }
    __syncthreads();
    for (int i = threadIdx.y; i < 32; i += 8) {
        out[(nBase + i) * K + (kBase + threadIdx.x)] = f32_to_bf16(tile[threadIdx.x][i]);
    }
}

// ---------------- shared GEMM core: 32(M) x 64(N) strip per wave, K=1024 ----------------
// Double-buffered in registers so next-chunk loads overlap current-chunk WMMAs.
static __device__ __forceinline__ void gemm_core_32x64(
        const unsigned short* __restrict__ A,   // bf16 [M][1024] row-major
        const unsigned short* __restrict__ B,   // bf16 [N][1024] row-major (transposed weight)
        int m0, int n0, int g, int mr, FragAcc acc[2][4]) {
    const unsigned short* arow0 = A + (size_t)(m0 + mr) * 1024;
    const unsigned short* arow1 = A + (size_t)(m0 + 16 + mr) * 1024;
    const unsigned short* brow0 = B + (size_t)(n0 +  0 + mr) * 1024;
    const unsigned short* brow1 = B + (size_t)(n0 + 16 + mr) * 1024;
    const unsigned short* brow2 = B + (size_t)(n0 + 32 + mr) * 1024;
    const unsigned short* brow3 = B + (size_t)(n0 + 48 + mr) * 1024;

    FragAB a0c, a1c, b0c, b1c, b2c, b3c;
    load_frag(a0c, arow0, 0, g);
    load_frag(a1c, arow1, 0, g);
    load_frag(b0c, brow0, 0, g);
    load_frag(b1c, brow1, 0, g);
    load_frag(b2c, brow2, 0, g);
    load_frag(b3c, brow3, 0, g);

#pragma unroll 2
    for (int kk = 0; kk < 1024 - 32; kk += 32) {
        FragAB a0n, a1n, b0n, b1n, b2n, b3n;
        int nx = kk + 32;
        load_frag(a0n, arow0, nx, g);
        load_frag(a1n, arow1, nx, g);
        load_frag(b0n, brow0, nx, g);
        load_frag(b1n, brow1, nx, g);
        load_frag(b2n, brow2, nx, g);
        load_frag(b3n, brow3, nx, g);

        acc[0][0].v = wmma_bf16(a0c, b0c, acc[0][0].v);
        acc[1][0].v = wmma_bf16(a1c, b0c, acc[1][0].v);
        acc[0][1].v = wmma_bf16(a0c, b1c, acc[0][1].v);
        acc[1][1].v = wmma_bf16(a1c, b1c, acc[1][1].v);
        acc[0][2].v = wmma_bf16(a0c, b2c, acc[0][2].v);
        acc[1][2].v = wmma_bf16(a1c, b2c, acc[1][2].v);
        acc[0][3].v = wmma_bf16(a0c, b3c, acc[0][3].v);
        acc[1][3].v = wmma_bf16(a1c, b3c, acc[1][3].v);

        a0c = a0n; a1c = a1n;
        b0c = b0n; b1c = b1n; b2c = b2n; b3c = b3n;
    }
    // last chunk
    acc[0][0].v = wmma_bf16(a0c, b0c, acc[0][0].v);
    acc[1][0].v = wmma_bf16(a1c, b0c, acc[1][0].v);
    acc[0][1].v = wmma_bf16(a0c, b1c, acc[0][1].v);
    acc[1][1].v = wmma_bf16(a1c, b1c, acc[1][1].v);
    acc[0][2].v = wmma_bf16(a0c, b2c, acc[0][2].v);
    acc[1][2].v = wmma_bf16(a1c, b2c, acc[1][2].v);
    acc[0][3].v = wmma_bf16(a0c, b3c, acc[0][3].v);
    acc[1][3].v = wmma_bf16(a1c, b3c, acc[1][3].v);
}

// ---------------- pass 1: QKV GEMM (x @ w_qkv + b), scatter to Q/K/Vt bf16 ----------------
// xb: bf16 [4096][1024]   wT: bf16 [3072][1024]
// Qb,Kb: bf16 [2][16][2048][64] (Q pre-scaled)   Vt: bf16 [2][16][64][2048]
__global__ void qkv_gemm_kernel(const unsigned short* __restrict__ xb,
                                const unsigned short* __restrict__ wT,
                                const float* __restrict__ bias,
                                unsigned short* __restrict__ Qb,
                                unsigned short* __restrict__ Kb,
                                unsigned short* __restrict__ Vt) {
    const int NSTRIPS = 3072 / 64;                    // 48
    int wave = blockIdx.x * 4 + (threadIdx.x >> 5);
    int lane = threadIdx.x & 31;
    int mt = wave / NSTRIPS, ns = wave % NSTRIPS;
    int m0 = mt * 32, n0 = ns * 64;
    int g = lane >> 4, mr = lane & 15;

    FragAcc acc[2][4] = {};
    gemm_core_32x64(xb, wT, m0, n0, g, mr, acc);

#pragma unroll
    for (int t = 0; t < 4; t++) {
        int col = n0 + t * 16 + mr;                   // 0..3071
        float bv = bias[col];
        int sec = col >> 10;                          // 0=Q 1=K 2=V
        int jj  = col & 1023;
        int h = jj >> 6, d = jj & 63;
#pragma unroll
        for (int mi = 0; mi < 2; mi++) {
#pragma unroll
            for (int r = 0; r < 8; r++) {
                int row = m0 + mi * 16 + r + 8 * g;   // 0..4095
                int bi = row >> 11, n = row & 2047;
                float val = acc[mi][t].f[r] + bv;
                if (sec == 0)
                    Qb[(((bi * 16 + h) * 2048 + n) << 6) + d] = f32_to_bf16(val * SCALE);
                else if (sec == 1)
                    Kb[(((bi * 16 + h) * 2048 + n) << 6) + d] = f32_to_bf16(val);
                else
                    Vt[(((bi * 16 + h) * 64 + d) << 11) + n] = f32_to_bf16(val);
            }
        }
    }
}

// ---------------- pass 2: attention, softmax over HEADS axis ----------------
// block = (batch, q-tile of 16); 16 waves = 16 heads; loop k in tiles of 32.
__global__ void attention_kernel(const unsigned short* __restrict__ Qb,
                                 const unsigned short* __restrict__ Kb,
                                 const unsigned short* __restrict__ Vt,
                                 unsigned short* __restrict__ wavg) {
    __shared__ __align__(16) float          dpS[16][32][16];   // [h][k][m]  32 KB
    __shared__ __align__(16) unsigned short atS[16][16][32];   // [h][m][k]  16 KB

    int h    = threadIdx.x >> 5;
    int lane = threadIdx.x & 31;
    int b  = blockIdx.x >> 7;
    int qt = blockIdx.x & 127;
    int q0 = qt * 16;
    int g = lane >> 4, mr = lane & 15;

    const unsigned short* Qh = Qb + (size_t)((b * 16 + h) * 2048) * 64;
    const unsigned short* Kh = Kb + (size_t)((b * 16 + h) * 2048) * 64;
    const unsigned short* Vh = Vt + (size_t)((b * 16 + h) * 64) * 2048;

    // Q A-fragments for this head's q-tile (d 0..31, 32..63); fixed for all iterations
    FragAB qa0, qa1;
    {
        const unsigned short* qrow = Qh + (q0 + mr) * 64;
        load_frag(qa0, qrow, 0, g);
        load_frag(qa1, qrow, 32, g);
    }

    FragAcc ctx[4] = {};           // 16 q x 64 d context accumulator

    int sm = threadIdx.x >> 5;     // softmax: my (m,k) position
    int sk = threadIdx.x & 31;

    for (int k0 = 0; k0 < 2048; k0 += 32) {
        // ---- issue ALL global loads for this iteration up front ----
        FragAB kb00, kb01, kb10, kb11;   // K^T B-frags: [kt][d-chunk]
        {
            const unsigned short* krow0 = Kh + (k0 + mr) * 64;
            const unsigned short* krow1 = Kh + (k0 + 16 + mr) * 64;
            load_frag(kb00, krow0, 0, g);
            load_frag(kb01, krow0, 32, g);
            load_frag(kb10, krow1, 0, g);
            load_frag(kb11, krow1, 32, g);
        }
        FragAB vb[4];                    // V B-frags, consumed only after softmax:
#pragma unroll                           // their loads overlap the LDS softmax phase
        for (int c = 0; c < 4; c++) {
            const unsigned short* vrow = Vh + (size_t)(c * 16 + mr) * 2048 + k0;
            load_frag(vb[c], vrow, 0, g);
        }

        // ---- dp = (Q*scale) @ K^T, two 16-wide k subtiles ----
        {
            FragAcc dp = {};
            dp.v = wmma_bf16(qa0, kb00, dp.v);
            dp.v = wmma_bf16(qa1, kb01, dp.v);
            float* p = &dpS[h][mr][8 * g];            // kt=0: 8 contiguous m values
            *(uint4*)(p)     = dp.u[0];
            *(uint4*)(p + 4) = dp.u[1];
        }
        {
            FragAcc dp = {};
            dp.v = wmma_bf16(qa0, kb10, dp.v);
            dp.v = wmma_bf16(qa1, kb11, dp.v);
            float* p = &dpS[h][16 + mr][8 * g];       // kt=1
            *(uint4*)(p)     = dp.u[0];
            *(uint4*)(p + 4) = dp.u[1];
        }
        __syncthreads();

        // ---- softmax across the 16 heads at my (m,k); write bf16 attn to [h][m][k] ----
        {
            float vals[16];
            float vmax = -1e30f;
#pragma unroll
            for (int hh = 0; hh < 16; hh++) {
                float v = dpS[hh][sk][sm];
                vals[hh] = v;
                vmax = fmaxf(vmax, v);
            }
            float s = 0.f;
#pragma unroll
            for (int hh = 0; hh < 16; hh++) {
                float e = __expf(vals[hh] - vmax);
                vals[hh] = e;
                s += e;
            }
            float inv = 1.0f / s;
#pragma unroll
            for (int hh = 0; hh < 16; hh++)
                atS[hh][sm][sk] = f32_to_bf16(vals[hh] * inv);
        }
        __syncthreads();

        // ---- ctx += attn(16x32) @ V(32x64); V frags already in registers ----
        FragAB at;
        load_frag(at, &atS[h][mr][0], 0, g);
#pragma unroll
        for (int c = 0; c < 4; c++)
            ctx[c].v = wmma_bf16(at, vb[c], ctx[c].v);
        // next iter's dpS writes are fenced by the next __syncthreads(); the atS
        // reads above complete before this wave signals it -> 2 barriers/iter.
    }

    // write context bf16 to wavg [b][n][h*64+d]
#pragma unroll
    for (int c = 0; c < 4; c++) {
        int d = c * 16 + mr;
#pragma unroll
        for (int r = 0; r < 8; r++) {
            int n = q0 + r + 8 * g;
            wavg[((size_t)(b * 2048 + n)) * 1024 + h * 64 + d] = f32_to_bf16(ctx[c].f[r]);
        }
    }
}

// ---------------- pass 3: output projection ----------------
// wavg bf16 [4096][1024] @ w_projT bf16 [1024][1024] + b_proj -> out fp32 [4096][1024]
__global__ void proj_gemm_kernel(const unsigned short* __restrict__ wavg,
                                 const unsigned short* __restrict__ wT,
                                 const float* __restrict__ bias,
                                 float* __restrict__ out) {
    int wave = blockIdx.x * 4 + (threadIdx.x >> 5);
    int lane = threadIdx.x & 31;
    int mt = wave >> 4, ns = wave & 15;               // 1024/64 = 16 strips
    int m0 = mt * 32, n0 = ns * 64;
    int g = lane >> 4, mr = lane & 15;

    FragAcc acc[2][4] = {};
    gemm_core_32x64(wavg, wT, m0, n0, g, mr, acc);

#pragma unroll
    for (int t = 0; t < 4; t++) {
        int col = n0 + t * 16 + mr;
        float bv = bias[col];
#pragma unroll
        for (int mi = 0; mi < 2; mi++) {
#pragma unroll
            for (int r = 0; r < 8; r++) {
                int row = m0 + mi * 16 + r + 8 * g;
                out[(size_t)row * 1024 + col] = acc[mi][t].f[r] + bv;
            }
        }
    }
}

// ---------------- launch ----------------

extern "C" void kernel_launch(void* const* d_in, const int* in_sizes, int n_in,
                              void* d_out, int out_size, void* d_ws, size_t ws_size,
                              hipStream_t stream) {
    (void)in_sizes; (void)n_in; (void)out_size; (void)ws_size;
    const float* x      = (const float*)d_in[0];   // (2,2048,1024)
    const float* w_qkv  = (const float*)d_in[1];   // (1024,3072)
    const float* b_qkv  = (const float*)d_in[2];   // (3072,)
    const float* w_proj = (const float*)d_in[3];   // (1024,1024)
    const float* b_proj = (const float*)d_in[4];   // (1024,)
    float* out = (float*)d_out;

    char* ws = (char*)d_ws;
    unsigned short* xb     = (unsigned short*)(ws);                 //  8.4 MB
    unsigned short* wqkvT  = (unsigned short*)(ws + 8388608);       //  6.3 MB
    unsigned short* wprojT = (unsigned short*)(ws + 14680064);      //  2.1 MB
    unsigned short* Qb     = (unsigned short*)(ws + 16777216);      //  8.4 MB
    unsigned short* Kb     = (unsigned short*)(ws + 25165824);      //  8.4 MB
    unsigned short* Vt     = (unsigned short*)(ws + 33554432);      //  8.4 MB
    unsigned short* wavg   = (unsigned short*)(ws + 41943040);      //  8.4 MB (total 48 MB)

    const int NX = 2 * 2048 * 1024;
    convert_x_kernel<<<(NX + 255) / 256, 256, 0, stream>>>(x, xb, NX);
    transpose_convert_kernel<<<dim3(3072 / 32, 1024 / 32), dim3(32, 8), 0, stream>>>(
        w_qkv, wqkvT, 1024, 3072);
    transpose_convert_kernel<<<dim3(1024 / 32, 1024 / 32), dim3(32, 8), 0, stream>>>(
        w_proj, wprojT, 1024, 1024);

    // 4096x3072 output, 32x64 strips per wave, 4 waves/block: 128*48/4 blocks
    qkv_gemm_kernel<<<(128 * 48) / 4, 128, 0, stream>>>(xb, wqkvT, b_qkv, Qb, Kb, Vt);

    // one block per (batch, 16-row q tile); 16 waves = 16 heads
    attention_kernel<<<2 * 128, 512, 0, stream>>>(Qb, Kb, Vt, wavg);

    // 4096x1024 output, 32x64 strips per wave: 128*16/4 blocks
    proj_gemm_kernel<<<(128 * 16) / 4, 128, 0, stream>>>(wavg, wprojT, b_proj, out);
}
